// MambaModel_87540023427863
// MI455X (gfx1250) — matrix operand order repaired and checked by reading, not compiled, not run on hardware
//
#include <hip/hip_runtime.h>
#include <hip/hip_bf16.h>
#include <stdint.h>

// ---------------------------------------------------------------------------
// Mamba LM forward on MI455X (gfx1250, wave32, WMMA).
// GEMMs: v_wmma_f32_16x16x32_bf16 with double-buffered LDS tiles filled by
// the Tensor Data Mover (tensor_load_to_lds, TENSORcnt), one DMA per tile,
// hardware LDS-padding and OOB zero-fill. DMA of tile k+1 overlaps WMMA on
// tile k.
// ---------------------------------------------------------------------------

#define V_  32000
#define DM_ 1024
#define DS_ 16
#define DC_ 4
#define DI_ 2048          // EXP * DM
#define DTR_ 64
#define B_  2
#define L_  2048
#define NTOK_ (B_ * L_)   // 4096

typedef __attribute__((ext_vector_type(16))) __bf16 v16bf;
typedef __attribute__((ext_vector_type(8)))  __bf16 v8bf;
typedef __attribute__((ext_vector_type(8)))  float  v8f;
typedef __attribute__((ext_vector_type(4)))  unsigned int v4u;
typedef __attribute__((ext_vector_type(4)))  int    v4i;
typedef __attribute__((ext_vector_type(8)))  int    v8i;

union V16U { v16bf v; v8bf h[2]; };

constexpr int BM = 128, BN = 128, BK = 32;
constexpr int LDT = BK + 8;     // padded LDS row stride (bf16 elems) = 40

__device__ __forceinline__ float softplus_f(float x) {
    return (x > 20.0f) ? x : log1pf(__expf(x));
}
__device__ __forceinline__ float silu_f(float x) {
    return x / (1.0f + __expf(-x));
}

// ---------------------------------------------------------------------------
// Tensor Data Mover: DMA a 2-D tile (tile_d1 rows x tile_d0 elems, bf16)
// from global to LDS.  LDS padding: after every 16 DWORDs (one 32-elem bf16
// row = 64B) insert 4 DWORDs (16B) -> row stride 40 bf16 == LDT.
// Rows >= tensor_d1 are zero-filled by hardware (OOB reads return zero).
// D# layout per CDNA5 ISA ch.8 (group0 128b, group1 256b, groups2/3 unused).
// ---------------------------------------------------------------------------
__device__ __forceinline__ void tdm_load_tile_2d(
    uint32_t lds_addr, const void* gaddr,
    uint32_t tensor_d0, uint32_t tensor_d1, uint32_t stride0,
    uint32_t tile_d0, uint32_t tile_d1)
{
    uint64_t ga = (uint64_t)(uintptr_t)gaddr;
    v4u g0;
    g0[0] = 1u;                                   // count=1, user mode
    g0[1] = lds_addr;                             // LDS byte address
    g0[2] = (uint32_t)ga;                         // global_addr[31:0]
    g0[3] = (uint32_t)(ga >> 32) | (2u << 30);    // global_addr[56:32] | type=2
    v8i g1;
    g1[0] = (int)((1u << 16)      // data_size = 1 (2 bytes)
                | (1u << 20)      // pad_enable
                | (3u << 22)      // pad_interval: 16 DWORDs between pads
                | (3u << 25));    // pad_amount: 4 DWORDs
    g1[1] = (int)((tensor_d0 & 0xffffu) << 16);                       // dim0 lo
    g1[2] = (int)(((tensor_d0 >> 16) & 0xffffu)
                | ((tensor_d1 & 0xffffu) << 16));                     // dim0 hi | dim1 lo
    g1[3] = (int)(((tensor_d1 >> 16) & 0xffffu) | (tile_d0 << 16));   // dim1 hi | tile0
    g1[4] = (int)(tile_d1 & 0xffffu);                                 // tile1 (tile2=0)
    g1[5] = (int)stride0;                                             // dim0 stride lo
    g1[6] = 0;                                                        // stride hi | d1str lo
    g1[7] = 0;
    v4i gz = (v4i){0, 0, 0, 0};
#if defined(__clang_major__) && (__clang_major__ >= 23)
    v8i gz8 = (v8i){0, 0, 0, 0, 0, 0, 0, 0};
    __builtin_amdgcn_tensor_load_to_lds(g0, g1, gz, gz, gz8, 0);
#else
    __builtin_amdgcn_tensor_load_to_lds(g0, g1, gz, gz, 0);
#endif
}

// ---------------------------------------------------------------------------
// fp32 -> bf16 streaming convert
// ---------------------------------------------------------------------------
__global__ void f2bf_kernel(const float* __restrict__ src,
                            __hip_bfloat16* __restrict__ dst, int n) {
    int i = blockIdx.x * blockDim.x + threadIdx.x;
    if (i < n) dst[i] = __float2bfloat16(src[i]);
}

// ---------------------------------------------------------------------------
// Embedding gather (fp32 table -> bf16 activations)
// ---------------------------------------------------------------------------
__global__ void gather_kernel(const int* __restrict__ tokens,
                              const float* __restrict__ embed,
                              __hip_bfloat16* __restrict__ embbf) {
    int idx = blockIdx.x * blockDim.x + threadIdx.x;   // over NTOK*DM
    if (idx >= NTOK_ * DM_) return;
    int c = idx & (DM_ - 1);
    int n = idx >> 10;                                  // DM_ == 1024
    int t = tokens[n];
    embbf[idx] = __float2bfloat16(embed[(size_t)t * DM_ + c]);
}

// ---------------------------------------------------------------------------
// Generic bf16 WMMA GEMM:  C[M x N] = A[M x K] * W[N x K]^T
//   EPI 0: store f32
//   EPI 1: store f32 of softplus(x + bias[col])
//   EPI 2: store f32 of (x + bias[col])
//   EPI 3: store bf16
// Block tile 128x128, BK=32, 256 threads = 8 waves, wave tile 64x32.
// Double-buffered LDS filled by TDM DMAs issued by wave 0 only; DMA of the
// next K-tile overlaps the current tile's WMMAs.  M % 128 == 0, K % 32 == 0
// at all call sites; N edge handled by TDM zero-fill + store masking.
// ---------------------------------------------------------------------------
template <int EPI>
__global__ __launch_bounds__(256)
void gemm_bf16_wmma(const __hip_bfloat16* __restrict__ A,
                    const __hip_bfloat16* __restrict__ W,
                    float* __restrict__ Cf,
                    __hip_bfloat16* __restrict__ Cbf,
                    const float* __restrict__ bias,
                    int M, int Nout, int K) {
    __shared__ __hip_bfloat16 As[2][BM * LDT];
    __shared__ __hip_bfloat16 Ws[2][BN * LDT];

    const int tid  = threadIdx.x;
    const int lane = tid & 31;
    const int wave = tid >> 5;
    const int wm   = wave & 1;                  // M half: 0/1 -> +0 / +64
    const int wn   = wave >> 1;                 // N quarter: *32
    const int blockN = blockIdx.x * BN;
    const int blockM = blockIdx.y * BM;

    const int halfSel = lane >> 4;              // 0: lanes 0-15, 1: lanes 16-31
    const int lr      = lane & 15;
    const int aKBase  = halfSel * 8;            // A frag: elems 0-7 -> K=aKBase..,
                                                //         elems 8-15 -> +16
    const int bKBase  = halfSel * 8;            // B frag: same K order as A frag

    // LDS byte addresses of the buffers (low 32 bits of flat address are the
    // LDS offset per the gfx1250 aperture rules).
    const uint32_t aBase[2] = { (uint32_t)(uintptr_t)&As[0][0],
                                (uint32_t)(uintptr_t)&As[1][0] };
    const uint32_t wBase[2] = { (uint32_t)(uintptr_t)&Ws[0][0],
                                (uint32_t)(uintptr_t)&Ws[1][0] };

    const uint32_t wAvail = (uint32_t)(Nout > blockN ? Nout - blockN : 0);

    auto stage_tile = [&](int buf, int k0) {
        // A tile: 128 rows x 32 K
        tdm_load_tile_2d(aBase[buf], A + (size_t)blockM * K + k0,
                         (uint32_t)(K - k0), (uint32_t)(M - blockM),
                         (uint32_t)K, BK, BM);
        // W tile: rows beyond Nout zero-filled by TDM OOB handling
        tdm_load_tile_2d(wBase[buf], W + (size_t)blockN * K + k0,
                         (uint32_t)(K - k0), wAvail,
                         (uint32_t)K, BK, BN);
    };

    v8f acc[4][2];
#pragma unroll
    for (int mi = 0; mi < 4; ++mi)
#pragma unroll
        for (int ni = 0; ni < 2; ++ni) acc[mi][ni] = (v8f){0,0,0,0,0,0,0,0};

    const int KT = K / BK;
    int buf = 0;
    if (wave == 0) {
        stage_tile(0, 0);
        __builtin_amdgcn_s_wait_tensorcnt(0);
    }
    __syncthreads();

    for (int kt = 0; kt < KT; ++kt) {
        // kick the DMA for the next K-tile into the other buffer
        if (wave == 0 && kt + 1 < KT) stage_tile(buf ^ 1, (kt + 1) * BK);

        const __hip_bfloat16* Ab = &As[buf][0];
        const __hip_bfloat16* Wb = &Ws[buf][0];

        v16bf afrag[4];
#pragma unroll
        for (int mi = 0; mi < 4; ++mi) {
            int r = wm * 64 + mi * 16 + lr;
            V16U t;
            t.h[0] = *(const v8bf*)(&Ab[r * LDT + aKBase]);
            t.h[1] = *(const v8bf*)(&Ab[r * LDT + aKBase + 16]);
            afrag[mi] = t.v;
        }
        v16bf bfrag[2];
#pragma unroll
        for (int ni = 0; ni < 2; ++ni) {
            int c = wn * 32 + ni * 16 + lr;
            V16U t;
            t.h[0] = *(const v8bf*)(&Wb[c * LDT + bKBase]);
            t.h[1] = *(const v8bf*)(&Wb[c * LDT + bKBase + 16]);
            bfrag[ni] = t.v;
        }

#pragma unroll
        for (int mi = 0; mi < 4; ++mi)
#pragma unroll
            for (int ni = 0; ni < 2; ++ni)
                acc[mi][ni] = __builtin_amdgcn_wmma_f32_16x16x32_bf16(
                    false, afrag[mi], false, bfrag[ni],
                    (short)0, acc[mi][ni], false, false);

        if (wave == 0) __builtin_amdgcn_s_wait_tensorcnt(0);  // next tile in LDS
        __syncthreads();
        buf ^= 1;
    }

    // --- epilogue (C/D layout: lane = col, VGPR r = row (+8 for hi half)) ---
#pragma unroll
    for (int mi = 0; mi < 4; ++mi) {
#pragma unroll
        for (int ni = 0; ni < 2; ++ni) {
            int col = blockN + wn * 32 + ni * 16 + lr;
            if (col >= Nout) continue;
#pragma unroll
            for (int r = 0; r < 8; ++r) {
                int row = blockM + wm * 64 + mi * 16 + (halfSel ? r + 8 : r);
                float v = acc[mi][ni][r];
                if (EPI == 1) v = softplus_f(v + bias[col]);
                if (EPI == 2) v = v + bias[col];
                size_t off = (size_t)row * Nout + col;
                if (EPI == 3) Cbf[off] = __float2bfloat16(v);
                else          Cf[off]  = v;
            }
        }
    }
}

// ---------------------------------------------------------------------------
// Causal depthwise conv (DC=4) + bias + SiLU. x comes from cols [0,DI) of xz.
// ---------------------------------------------------------------------------
__global__ void conv_silu_kernel(const float* __restrict__ xz,
                                 const float* __restrict__ conv_w,
                                 const float* __restrict__ conv_b,
                                 float* __restrict__ xact,
                                 __hip_bfloat16* __restrict__ xbf) {
    int idx = blockIdx.x * blockDim.x + threadIdx.x;    // over B*L*DI
    if (idx >= B_ * L_ * DI_) return;
    int d = idx % DI_;
    int l = (idx / DI_) % L_;
    int b = idx / (DI_ * L_);
    float acc = conv_b[d];
#pragma unroll
    for (int j = 0; j < DC_; ++j) {
        int ls = l - (DC_ - 1) + j;
        if (ls >= 0)
            acc += xz[(size_t)(b * L_ + ls) * (2 * DI_) + d] * conv_w[d * DC_ + j];
    }
    float v = silu_f(acc);
    xact[idx] = v;
    xbf[idx]  = __float2bfloat16(v);
}

// ---------------------------------------------------------------------------
// Slice dt (first DTR cols of dbc) to bf16 for the dt_proj GEMM.
// ---------------------------------------------------------------------------
__global__ void slice_dt_kernel(const float* __restrict__ dbc,
                                __hip_bfloat16* __restrict__ dtbf) {
    int idx = blockIdx.x * blockDim.x + threadIdx.x;    // over NTOK*DTR
    if (idx >= NTOK_ * DTR_) return;
    int r = idx % DTR_;
    int n = idx / DTR_;
    dtbf[idx] = __float2bfloat16(dbc[(size_t)n * 96 + r]);
}

// ---------------------------------------------------------------------------
// Selective scan: one thread per (b, d). 16-state recurrence in registers,
// serial over L. Fuses y = (scan + x*D) * silu(z), emits bf16 for out_proj.
// global_prefetch lookahead hides the per-step latency of dtq/xact streams.
// ---------------------------------------------------------------------------
__global__ __launch_bounds__(256)
void scan_kernel(const float* __restrict__ dtq,     // NTOK x DI (softplus'd)
                 const float* __restrict__ dbc,     // NTOK x 96
                 const float* __restrict__ xact,    // NTOK x DI
                 const float* __restrict__ xz,      // NTOK x 2DI (z at +DI)
                 const float* __restrict__ A_log,   // DI x DS
                 const float* __restrict__ Dp,      // DI
                 __hip_bfloat16* __restrict__ ybf)  // NTOK x DI
{
    int d = blockIdx.x * blockDim.x + threadIdx.x;
    int b = blockIdx.y;
    if (d >= DI_) return;

    float An[DS_];
#pragma unroll
    for (int s = 0; s < DS_; ++s) An[s] = -__expf(A_log[(size_t)d * DS_ + s]);
    const float Dd = Dp[d];

    float h[DS_];
#pragma unroll
    for (int s = 0; s < DS_; ++s) h[s] = 0.0f;

    constexpr int PF = 8;   // prefetch lookahead (steps)
    for (int l = 0; l < L_; ++l) {
        size_t n = (size_t)b * L_ + l;
        if (l + PF < L_) {
            size_t np = n + PF;
            __builtin_prefetch(dtq  + np * DI_ + d, 0, 0);
            __builtin_prefetch(xact + np * DI_ + d, 0, 0);
            __builtin_prefetch(xz   + np * (2 * DI_) + DI_ + d, 0, 0);
        }
        float dt = dtq[n * DI_ + d];
        float u  = xact[n * DI_ + d];
        const float4* bc4 = (const float4*)(dbc + n * 96 + DTR_);
        float du = dt * u;
        float y = 0.0f;
#pragma unroll
        for (int q = 0; q < 4; ++q) {
            float4 Bq = bc4[q];        // B_t[4q..4q+3]
            float4 Cq = bc4[q + 4];    // C_t[4q..4q+3]
            float e;
            e = __expf(dt * An[4*q+0]); h[4*q+0] = e*h[4*q+0] + du*Bq.x; y += h[4*q+0]*Cq.x;
            e = __expf(dt * An[4*q+1]); h[4*q+1] = e*h[4*q+1] + du*Bq.y; y += h[4*q+1]*Cq.y;
            e = __expf(dt * An[4*q+2]); h[4*q+2] = e*h[4*q+2] + du*Bq.z; y += h[4*q+2]*Cq.z;
            e = __expf(dt * An[4*q+3]); h[4*q+3] = e*h[4*q+3] + du*Bq.w; y += h[4*q+3]*Cq.w;
        }
        float zv = xz[n * (2 * DI_) + DI_ + d];
        float out = (y + u * Dd) * silu_f(zv);
        ybf[n * DI_ + d] = __float2bfloat16(out);
    }
}

// ---------------------------------------------------------------------------
// Host launcher
// ---------------------------------------------------------------------------
extern "C" void kernel_launch(void* const* d_in, const int* in_sizes, int n_in,
                              void* d_out, int out_size, void* d_ws, size_t ws_size,
                              hipStream_t stream) {
    const int*   tokens    = (const int*)  d_in[0];
    const float* embed     = (const float*)d_in[1];
    const float* in_proj_w = (const float*)d_in[2];
    const float* conv_w    = (const float*)d_in[3];
    const float* conv_b    = (const float*)d_in[4];
    const float* x_proj_w  = (const float*)d_in[5];
    const float* dt_proj_w = (const float*)d_in[6];
    const float* dt_proj_b = (const float*)d_in[7];
    const float* A_log     = (const float*)d_in[8];
    const float* Dp        = (const float*)d_in[9];
    const float* out_proj_w= (const float*)d_in[10];
    const float* head_w    = (const float*)d_in[11];
    const float* head_b    = (const float*)d_in[12];
    float* out = (float*)d_out;

    char* ws = (char*)d_ws;
    size_t off = 0;
    auto alloc = [&](size_t bytes) -> void* {
        void* p = ws + off;
        off += (bytes + 255) & ~(size_t)255;
        return p;
    };

    __hip_bfloat16* w_inproj  = (__hip_bfloat16*)alloc((size_t)2*DI_*DM_ * 2);
    __hip_bfloat16* w_xproj   = (__hip_bfloat16*)alloc((size_t)96*DI_ * 2);
    __hip_bfloat16* w_dtproj  = (__hip_bfloat16*)alloc((size_t)DI_*DTR_ * 2);
    __hip_bfloat16* w_outproj = (__hip_bfloat16*)alloc((size_t)DM_*DI_ * 2);
    __hip_bfloat16* w_head    = (__hip_bfloat16*)alloc((size_t)V_*DM_ * 2);
    __hip_bfloat16* emb_bf    = (__hip_bfloat16*)alloc((size_t)NTOK_*DM_ * 2);
    float*          xz        = (float*)        alloc((size_t)NTOK_*2*DI_ * 4);
    float*          xact      = (float*)        alloc((size_t)NTOK_*DI_ * 4);
    __hip_bfloat16* xbf       = (__hip_bfloat16*)alloc((size_t)NTOK_*DI_ * 2);
    float*          dbc       = (float*)        alloc((size_t)NTOK_*96 * 4);
    __hip_bfloat16* dtbf      = (__hip_bfloat16*)alloc((size_t)NTOK_*DTR_ * 2);
    float*          dtq       = (float*)        alloc((size_t)NTOK_*DI_ * 4);
    __hip_bfloat16* ybf       = (__hip_bfloat16*)alloc((size_t)NTOK_*DI_ * 2);
    __hip_bfloat16* y2bf      = (__hip_bfloat16*)alloc((size_t)NTOK_*DM_ * 2);
    (void)ws_size; (void)in_sizes; (void)n_in; (void)out_size;

    const int T = 256;
    auto cdiv = [](int a, int b) { return (a + b - 1) / b; };

    // 1) weight fp32 -> bf16 conversions
    {
        struct { const float* s; __hip_bfloat16* d; int n; } cv[5] = {
            { in_proj_w,  w_inproj,  2*DI_*DM_ },
            { x_proj_w,   w_xproj,   96*DI_    },
            { dt_proj_w,  w_dtproj,  DI_*DTR_  },
            { out_proj_w, w_outproj, DM_*DI_   },
            { head_w,     w_head,    V_*DM_    },
        };
        for (int i = 0; i < 5; ++i)
            f2bf_kernel<<<cdiv(cv[i].n, T), T, 0, stream>>>(cv[i].s, cv[i].d, cv[i].n);
    }

    // 2) embedding gather -> bf16
    gather_kernel<<<cdiv(NTOK_*DM_, T), T, 0, stream>>>(tokens, embed, emb_bf);

    // 3) in_proj: xz = emb @ in_proj_w^T   (4096 x 4096 x 1024)
    gemm_bf16_wmma<0><<<dim3(cdiv(2*DI_,128), NTOK_/128), T, 0, stream>>>(
        emb_bf, w_inproj, xz, nullptr, nullptr, NTOK_, 2*DI_, DM_);

    // 4) depthwise conv + SiLU
    conv_silu_kernel<<<cdiv(NTOK_*DI_, T), T, 0, stream>>>(
        xz, conv_w, conv_b, xact, xbf);

    // 5) x_proj: dbc = x @ x_proj_w^T   (4096 x 96 x 2048)
    gemm_bf16_wmma<0><<<dim3(cdiv(96,128), NTOK_/128), T, 0, stream>>>(
        xbf, w_xproj, dbc, nullptr, nullptr, NTOK_, 96, DI_);

    // 6) slice dt columns -> bf16
    slice_dt_kernel<<<cdiv(NTOK_*DTR_, T), T, 0, stream>>>(dbc, dtbf);

    // 7) dt_proj + softplus: dtq = softplus(dt @ dt_proj_w^T + b)  (4096x2048x64)
    gemm_bf16_wmma<1><<<dim3(cdiv(DI_,128), NTOK_/128), T, 0, stream>>>(
        dtbf, w_dtproj, dtq, nullptr, dt_proj_b, NTOK_, DI_, DTR_);

    // 8) selective scan + gating -> ybf
    scan_kernel<<<dim3(DI_/T, B_), T, 0, stream>>>(
        dtq, dbc, xact, xz, A_log, Dp, ybf);

    // 9) out_proj: y2 = y @ out_proj_w^T -> bf16   (4096 x 1024 x 2048)
    gemm_bf16_wmma<3><<<dim3(cdiv(DM_,128), NTOK_/128), T, 0, stream>>>(
        ybf, w_outproj, nullptr, y2bf, nullptr, NTOK_, DM_, DI_);

    // 10) head: logits = y2 @ head_w^T + head_b   (4096 x 32000 x 1024)
    gemm_bf16_wmma<2><<<dim3(cdiv(V_,128), NTOK_/128), T, 0, stream>>>(
        y2bf, w_head, out, nullptr, head_b, NTOK_, V_, DM_);
}
